// PAB_79207786873124
// MI455X (gfx1250) — matrix operand order repaired and checked by reading, not compiled, not run on hardware
//
#include <hip/hip_runtime.h>
#include <hip/hip_bf16.h>

typedef __bf16 bf16_t;
typedef bf16_t v16bf __attribute__((ext_vector_type(16)));
typedef float  v8f   __attribute__((ext_vector_type(8)));

#define C_DIM 256
#define N_DIM 2304            // 48*48
#define B_DIM 8
#define FEA_SZ (C_DIM * N_DIM)  // 589824

#define XT_STR 40             // X-transposed LDS row stride (bf16)
#define KSTR   264            // K tile LDS row stride (bf16), 528B rows
#define VSTR   40             // V-transposed LDS row stride (bf16), 80B rows
#define PSTR   40             // P tile LDS row stride (bf16)
#define OSTR   68             // O transpose staging stride (f32)

union FragAB { v16bf v; uint4 u[2]; };

// ---------------------------------------------------------------------------
// Fold BN into conv1x1: W'[o,c] = W[o,c]*g[o]/sqrt(v[o]+eps) (as bf16)
//                       b'[o]   = (b[o]-m[o])*scale + beta[o] (as f32)
// ---------------------------------------------------------------------------
__global__ void fold_kernel(
    const float* Wq, const float* bq, const float* gq, const float* betaq,
    const float* mq, const float* vq,
    const float* Wk, const float* bk, const float* gk, const float* betak,
    const float* mk, const float* vk,
    bf16_t* Wfold, float* bfold)
{
    int id = blockIdx.x * blockDim.x + threadIdx.x;
    if (id >= 2 * C_DIM * C_DIM) return;
    int sel = id >> 16;           // 0 = query branch, 1 = key branch
    int rem = id & 0xFFFF;
    int o = rem >> 8, c = rem & 255;
    const float* W  = sel ? Wk : Wq;
    const float* bb = sel ? bk : bq;
    const float* gg = sel ? gk : gq;
    const float* be = sel ? betak : betaq;
    const float* mm = sel ? mk : mq;
    const float* vv = sel ? vk : vq;
    float scale = gg[o] * rsqrtf(vv[o] + 1e-5f);
    Wfold[(size_t)sel * C_DIM * C_DIM + o * C_DIM + c] =
        (bf16_t)(W[o * C_DIM + c] * scale);
    if (c == 0)
        bfold[sel * C_DIM + o] = (bb[o] - mm[o]) * scale + be[o];
}

// ---------------------------------------------------------------------------
// Projection GEMM: P = W'(256x256) * X(256x2304) + b', stored bf16 in the
// natural (C,N) row-major layout. The (N,C) Q/K view is the same buffer.
// Block: 64 p-columns, all 256 o-rows (4 waves x 64 o). K loop in chunks of 32.
// ---------------------------------------------------------------------------
__global__ void __launch_bounds__(128) proj_kernel(
    const float* fea_left, const float* fea_right,
    const bf16_t* Wfold, const float* bfold, bf16_t* Pbuf)
{
    const int tid  = threadIdx.x;
    const int lane = tid & 31, wave = tid >> 5;
    const int half = lane >> 4, l16 = lane & 15;
    const int p0 = blockIdx.x * 64;
    const int b  = blockIdx.y;
    const int z  = blockIdx.z;   // 0:Q(left) 1:K(right) 2:Q(right) 3:K(left)
    const int wsel = (z == 1 || z == 3);
    const float* src = ((z == 0 || z == 3) ? fea_left : fea_right) + (size_t)b * FEA_SZ;
    const bf16_t* Wp = Wfold + (size_t)wsel * C_DIM * C_DIM;
    const float*  bp = bfold + wsel * C_DIM;
    bf16_t* Pout = Pbuf + ((size_t)z * B_DIM + b) * FEA_SZ;

    __shared__ bf16_t Xt[64 * XT_STR];   // transposed X tile: [p][c], padded

    v8f acc[4][4];
    const v8f vzero = {0.f, 0.f, 0.f, 0.f, 0.f, 0.f, 0.f, 0.f};
#pragma unroll
    for (int i = 0; i < 4; ++i)
#pragma unroll
        for (int j = 0; j < 4; ++j) acc[i][j] = vzero;

    const int o0 = wave * 64;

    for (int kc = 0; kc < 8; ++kc) {
        const int c0 = kc * 32;
        __syncthreads();
        // stage X chunk transposed: LDS[p][c] = (bf16)X[c][p]
        for (int e = tid; e < 32 * 64; e += 128) {
            int ci = e >> 6, pi = e & 63;
            Xt[pi * XT_STR + ci] = (bf16_t)src[(size_t)(c0 + ci) * N_DIM + p0 + pi];
        }
        __syncthreads();

        // A fragments from folded weights (row-major, K pairs contiguous)
        FragAB afr[4];
#pragma unroll
        for (int of = 0; of < 4; ++of) {
            const bf16_t* rp = Wp + (size_t)(o0 + of * 16 + l16) * C_DIM + c0 + half * 8;
            afr[of].u[0] = *(const uint4*)rp;        // K = base8 .. base8+7
            afr[of].u[1] = *(const uint4*)(rp + 16); // K = 16+base8 .. +7
        }
#pragma unroll
        for (int pf = 0; pf < 4; ++pf) {
            FragAB bfr;   // B[k=c, n=p] from transposed tile (k contiguous)
            const bf16_t* rp = &Xt[(pf * 16 + l16) * XT_STR + half * 8];
            bfr.u[0] = *(const uint4*)rp;
            bfr.u[1] = *(const uint4*)(rp + 16);
#pragma unroll
            for (int of = 0; of < 4; ++of)
                acc[of][pf] = __builtin_amdgcn_wmma_f32_16x16x32_bf16(
                    false, afr[of].v, false, bfr.v, (short)0, acc[of][pf],
                    false, false);
        }
    }

    // add bias, convert, store (D layout: row = v + 8*half, col = l16)
#pragma unroll
    for (int of = 0; of < 4; ++of)
#pragma unroll
        for (int pf = 0; pf < 4; ++pf)
#pragma unroll
            for (int v = 0; v < 8; ++v) {
                int o = o0 + of * 16 + v + 8 * half;
                int p = p0 + pf * 16 + l16;
                Pout[(size_t)o * N_DIM + p] = (bf16_t)(acc[of][pf][v] + bp[o]);
            }
}

// ---------------------------------------------------------------------------
// Flash attention: O = softmax(Q K^T / 256) V, per (batch, direction).
// Block = 64 Q rows (16 per wave, Q held in registers), key blocks of 32.
// Output written transposed (channel-major) for the (b,c,h,w) result.
// ---------------------------------------------------------------------------
__global__ void __launch_bounds__(128) attn_kernel(
    const float* fea_left, const float* fea_right,
    const bf16_t* Pbuf, float* out)
{
    const int tid  = threadIdx.x;
    const int lane = tid & 31, wave = tid >> 5;
    const int half = lane >> 4, l16 = lane & 15;
    const int n0  = blockIdx.x * 64;
    const int b   = blockIdx.y;
    const int dir = blockIdx.z;

    const bf16_t* Qm = Pbuf + ((size_t)(dir ? 2 : 0) * B_DIM + b) * FEA_SZ;
    const bf16_t* Km = Pbuf + ((size_t)(dir ? 3 : 1) * B_DIM + b) * FEA_SZ;
    const float*  Vm = (dir ? fea_left : fea_right) + (size_t)b * FEA_SZ;
    float* outp = out + ((size_t)dir * B_DIM + b) * FEA_SZ;

    __shared__ union {
        struct {
            bf16_t K [32 * KSTR];        // key tile, row-major [key][d]
            bf16_t Vt[C_DIM * VSTR];     // value tile transposed [d][key]
            bf16_t P [4 * 16 * PSTR];    // per-wave prob tiles [qrow][key]
        } s;
        float O[64 * OSTR];              // output transpose staging
    } L;

    // preload this wave's 16 Q rows as 8 A-fragments (d chunks of 32)
    FragAB qfr[8];
    const size_t qrow = (size_t)(n0 + wave * 16 + l16);
#pragma unroll
    for (int kc = 0; kc < 8; ++kc) {
        const bf16_t* rp = Qm + qrow * C_DIM + kc * 32 + half * 8;
        qfr[kc].u[0] = *(const uint4*)rp;
        qfr[kc].u[1] = *(const uint4*)(rp + 16);
    }

    const v8f vzero = {0.f, 0.f, 0.f, 0.f, 0.f, 0.f, 0.f, 0.f};
    v8f acc[16];
#pragma unroll
    for (int f = 0; f < 16; ++f) acc[f] = vzero;
    float mrow[8], lrow[8];
#pragma unroll
    for (int v = 0; v < 8; ++v) { mrow[v] = -3.0e38f; lrow[v] = 0.f; }

    bf16_t* Pw = &L.s.P[wave * 16 * PSTR];

    for (int kb = 0; kb < N_DIM / 32; ++kb) {
        __syncthreads();
        // stage K tile (32 keys x 256 d, bf16) — 16B vector copies
        for (int e = tid; e < 32 * 32; e += 128) {
            int r = e >> 5, c8 = e & 31;
            *(uint4*)&L.s.K[r * KSTR + c8 * 8] =
                *(const uint4*)(Km + (size_t)(kb * 32 + r) * C_DIM + c8 * 8);
        }
        // stage V tile transposed, fp32 -> bf16 on the fly
        for (int e = tid; e < 32 * 64; e += 128) {
            int r = e >> 6, c4 = e & 63;
            float4 vv = *(const float4*)(Vm + (size_t)(kb * 32 + r) * C_DIM + c4 * 4);
            bf16_t* dst = &L.s.Vt[(c4 * 4) * VSTR + r];
            dst[0 * VSTR] = (bf16_t)vv.x; dst[1 * VSTR] = (bf16_t)vv.y;
            dst[2 * VSTR] = (bf16_t)vv.z; dst[3 * VSTR] = (bf16_t)vv.w;
        }
        __syncthreads();

        // S = Q K^T (two 16x16 n-fragments, 8 k-chunks of 32)
        v8f s[2];
        s[0] = vzero; s[1] = vzero;
#pragma unroll
        for (int kc = 0; kc < 8; ++kc)
#pragma unroll
            for (int j = 0; j < 2; ++j) {
                FragAB bfr;  // B[k=d, n=key] : contiguous d in K-tile row
                const bf16_t* rp = &L.s.K[(j * 16 + l16) * KSTR + kc * 32 + half * 8];
                bfr.u[0] = *(const uint4*)rp;
                bfr.u[1] = *(const uint4*)(rp + 16);
                s[j] = __builtin_amdgcn_wmma_f32_16x16x32_bf16(
                    false, qfr[kc].v, false, bfr.v, (short)0, s[j], false, false);
            }
        s[0] *= (1.f / 256.f);
        s[1] *= (1.f / 256.f);

        // online softmax; D-layout row = v + 8*half, cols across 16 lanes
        float alpha[8];
#pragma unroll
        for (int v = 0; v < 8; ++v) {
            float rm = fmaxf(s[0][v], s[1][v]);
#pragma unroll
            for (int mk = 1; mk < 16; mk <<= 1)
                rm = fmaxf(rm, __shfl_xor(rm, mk, 32));
            float mnew = fmaxf(mrow[v], rm);
            float a = __expf(mrow[v] - mnew);
            mrow[v] = mnew;
            float p0 = __expf(s[0][v] - mnew);
            float p1 = __expf(s[1][v] - mnew);
            s[0][v] = p0; s[1][v] = p1;
            float rs = p0 + p1;
#pragma unroll
            for (int mk = 1; mk < 16; mk <<= 1)
                rs += __shfl_xor(rs, mk, 32);
            lrow[v] = lrow[v] * a + rs;
            alpha[v] = a;
        }
        v8f av;
#pragma unroll
        for (int v = 0; v < 8; ++v) av[v] = alpha[v];
#pragma unroll
        for (int f = 0; f < 16; ++f) acc[f] *= av;

        // route P through per-wave LDS: D layout -> A layout
#pragma unroll
        for (int j = 0; j < 2; ++j)
#pragma unroll
            for (int v = 0; v < 8; ++v)
                Pw[(v + 8 * half) * PSTR + j * 16 + l16] = (bf16_t)s[j][v];
        asm volatile("s_wait_dscnt 0" ::: "memory");
        __builtin_amdgcn_wave_barrier();

        FragAB pfr;
        {
            const bf16_t* rp = Pw + l16 * PSTR + half * 8;
            pfr.u[0] = *(const uint4*)rp;
            pfr.u[1] = *(const uint4*)(rp + 16);
        }
        // O += P V (16 output-column fragments, one 32-key chunk)
#pragma unroll
        for (int f = 0; f < 16; ++f) {
            FragAB bfr;  // B[k=key, n=d] from transposed V tile
            const bf16_t* rp = &L.s.Vt[(f * 16 + l16) * VSTR + half * 8];
            bfr.u[0] = *(const uint4*)rp;
            bfr.u[1] = *(const uint4*)(rp + 16);
            acc[f] = __builtin_amdgcn_wmma_f32_16x16x32_bf16(
                false, pfr.v, false, bfr.v, (short)0, acc[f], false, false);
        }
    }

    // normalize by row sums
    v8f linv;
#pragma unroll
    for (int v = 0; v < 8; ++v) linv[v] = 1.f / lrow[v];
#pragma unroll
    for (int f = 0; f < 16; ++f) acc[f] *= linv;

    // transpose through LDS, 64 channels per pass, coalesced float4 stores
    for (int g = 0; g < 4; ++g) {
        __syncthreads();
#pragma unroll
        for (int ff = 0; ff < 4; ++ff) {
            int f = g * 4 + ff;
#pragma unroll
            for (int v = 0; v < 8; ++v) {
                int dloc = ff * 16 + l16;
                int nloc = wave * 16 + v + 8 * half;
                L.O[dloc * OSTR + nloc] = acc[f][v];
            }
        }
        __syncthreads();
        for (int r = tid; r < 64 * 4; r += 128) {
            int dloc = r >> 2, q = r & 3;
            int ch = g * 64 + dloc;
            const float* srcp = &L.O[dloc * OSTR + q * 16];
            float4* dst = (float4*)(outp + (size_t)ch * N_DIM + n0 + q * 16);
#pragma unroll
            for (int t = 0; t < 4; ++t) dst[t] = *(const float4*)(srcp + 4 * t);
        }
    }
}

// ---------------------------------------------------------------------------
extern "C" void kernel_launch(void* const* d_in, const int* in_sizes, int n_in,
                              void* d_out, int out_size, void* d_ws, size_t ws_size,
                              hipStream_t stream)
{
    const float* fea_left  = (const float*)d_in[0];
    const float* fea_right = (const float*)d_in[1];
    const float* Wq = (const float*)d_in[2];  const float* bq = (const float*)d_in[3];
    const float* gq = (const float*)d_in[4];  const float* betaq = (const float*)d_in[5];
    const float* mq = (const float*)d_in[6];  const float* vq = (const float*)d_in[7];
    const float* Wk = (const float*)d_in[8];  const float* bk = (const float*)d_in[9];
    const float* gk = (const float*)d_in[10]; const float* betak = (const float*)d_in[11];
    const float* mk = (const float*)d_in[12]; const float* vk = (const float*)d_in[13];

    char* ws = (char*)d_ws;
    bf16_t* Wfold = (bf16_t*)ws;                               // 2*256*256 bf16
    float*  bfold = (float*)(ws + 2 * C_DIM * C_DIM * 2);      // 512 f32
    bf16_t* Pbuf  = (bf16_t*)(ws + 266240);                    // 4*8*589824 bf16

    fold_kernel<<<dim3(1024), dim3(128), 0, stream>>>(
        Wq, bq, gq, betaq, mq, vq, Wk, bk, gk, betak, mk, vk, Wfold, bfold);

    proj_kernel<<<dim3(36, 8, 4), dim3(128), 0, stream>>>(
        fea_left, fea_right, Wfold, bfold, Pbuf);

    attn_kernel<<<dim3(36, 8, 2), dim3(128), 0, stream>>>(
        fea_left, fea_right, Pbuf, (float*)d_out);
}
